// Lstm_68942815036141
// MI455X (gfx1250) — matrix operand order repaired and checked by reading, not compiled, not run on hardware
//
#include <hip/hip_runtime.h>
#include <hip/hip_bf16.h>
#include <cstddef>

typedef __bf16 bf16_t;
typedef __attribute__((ext_vector_type(8)))  __bf16 v8bf;
typedef __attribute__((ext_vector_type(16))) __bf16 v16bf;
typedef __attribute__((ext_vector_type(8)))  float  v8f;

#define T_STEPS 256
#define BATCH   64
#define VOCABSZ 512
#define EMBSZ   256
#define NN0     1536
#define NN1     1536

// ---------------------------------------------------------------------------
// f32 -> bf16 bulk convert
// ---------------------------------------------------------------------------
__global__ void f32_to_bf16_kernel(const float* __restrict__ src,
                                   bf16_t* __restrict__ dst, int n) {
    int i = blockIdx.x * blockDim.x + threadIdx.x;
    if (i < n) dst[i] = (bf16_t)src[i];
}

// ---------------------------------------------------------------------------
// Pack a row-major fp32 K x N matrix into WMMA B-fragment-major bf16 tiles.
// Tile (kt, nt) covers K rows [kt*32, kt*32+32), N cols [nt*16, nt*16+16).
// Within a tile: 512 bf16 stored lane-major (32 lanes x 16 elems); lane l
// (n = l&15, khalf = l>>4) holds K = kt*32 + (e>>3)*16 + khalf*8 + (e&7) (e = 0..15),
// matching the 16-bit 32x16 B operand layout (cf. sparse-B layout, 05_wmma).
// Tiles ordered nt-major so the GEMM k-loop streams contiguously.
// ---------------------------------------------------------------------------
__global__ void pack_b_kernel(const float* __restrict__ src,
                              bf16_t* __restrict__ dst, int K, int N) {
    int idx = blockIdx.x * blockDim.x + threadIdx.x;
    if (idx >= K * N) return;
    int KT   = K >> 5;
    int tile = idx >> 9;      // / 512
    int r    = idx & 511;
    int lane = r >> 4;
    int e    = r & 15;
    int nt   = tile / KT;
    int kt   = tile - nt * KT;
    int k = kt * 32 + ((e >> 3) << 4) + ((lane >> 4) << 3) + (e & 7);
    int n = nt * 16 + (lane & 15);
    dst[idx] = (bf16_t)src[(size_t)k * N + n];
}

// ---------------------------------------------------------------------------
// A-fragment loader: 16-bit A 16x32 layout (05_wmma 7.12.2):
//   lanes 0-15 : M=lane,    VGPR0-3 K=0..7,   VGPR4-7 K=16..23
//   lanes 16-31: M=lane-16, VGPR0-3 K=8..15,  VGPR4-7 K=24..31
// => per lane two contiguous 8-element (16 B) chunks.
// ---------------------------------------------------------------------------
__device__ __forceinline__ v16bf load_a_frag(const bf16_t* __restrict__ A,
                                             int lda, int row0, int k0) {
    int l = threadIdx.x & 31;
    const bf16_t* p = A + (size_t)(row0 + (l & 15)) * lda + k0 + ((l >> 4) << 3);
    v8bf lo = *(const v8bf*)p;
    v8bf hi = *(const v8bf*)(p + 16);
    v16bf r;
#pragma unroll
    for (int i = 0; i < 8; ++i) { r[i] = lo[i]; r[i + 8] = hi[i]; }
    return r;
}

// ---------------------------------------------------------------------------
// WMMA GEMM:  C(MxN) = [A0 | A1](Mx(K0+K1)) * Bpacked + bias
// One wave computes (MT*16) x (NT*16); k-loop is software-pipelined 2 deep
// (ping-pong fragment registers) so next-tile loads overlap current WMMAs.
// __launch_bounds__(BLOCK, 1) lets the allocator use the full 256-VGPR file
// (MT=2,NT=2 double-buffered ~= 130 live VGPRs -> no scratch spills).
// Requires: K0,K1 mult of 32; (K0+K1)/32 even; M mult of MT*16; N mult NT*16.
// ---------------------------------------------------------------------------
template <int BLOCK, int MT, int NT, bool STORE_BF16>
__global__ __launch_bounds__(BLOCK, 1)
void wmma_gemm_kernel(const bf16_t* __restrict__ A0, int lda0, int K0,
                      const bf16_t* __restrict__ A1, int lda1, int K1,
                      const bf16_t* __restrict__ Bp, int N,
                      const float* __restrict__ bias,
                      void* __restrict__ Cout, int ldc, int M) {
    const int lane = threadIdx.x & 31;
    int wave    = blockIdx.x * (BLOCK >> 5) + (threadIdx.x >> 5);
    int nGroups = (N >> 4) / NT;
    int mg = wave / nGroups;
    int ng = wave - mg * nGroups;
    if (mg * MT * 16 >= M) return;

    const int KT  = (K0 + K1) >> 5;
    const int KT0 = K0 >> 5;

    v8f acc[MT][NT];
#pragma unroll
    for (int i = 0; i < MT; ++i)
#pragma unroll
        for (int j = 0; j < NT; ++j)
#pragma unroll
            for (int r = 0; r < 8; ++r) acc[i][j][r] = 0.0f;

    v16bf aC[MT], bC[NT], aN[MT], bN[NT];

    auto loadA = [&](v16bf (&dst)[MT], int kt) {
#pragma unroll
        for (int i = 0; i < MT; ++i)
            dst[i] = (kt < KT0)
                         ? load_a_frag(A0, lda0, (mg * MT + i) * 16, kt * 32)
                         : load_a_frag(A1, lda1, (mg * MT + i) * 16, (kt - KT0) * 32);
    };
    auto loadB = [&](v16bf (&dst)[NT], int kt) {
#pragma unroll
        for (int j = 0; j < NT; ++j) {
            const bf16_t* tb =
                Bp + (((size_t)(ng * NT + j) * KT + kt) << 9) + (lane << 4);
            dst[j] = *(const v16bf*)tb;
        }
    };
    auto mmas = [&](v16bf (&a)[MT], v16bf (&b)[NT]) {
#pragma unroll
        for (int i = 0; i < MT; ++i)
#pragma unroll
            for (int j = 0; j < NT; ++j)
                acc[i][j] = __builtin_amdgcn_wmma_f32_16x16x32_bf16(
                    false, a[i], false, b[j], (short)0, acc[i][j], false, false);
    };

    loadA(aC, 0);
    loadB(bC, 0);
    for (int kt = 0; kt < KT; kt += 2) {
        loadA(aN, kt + 1);          // prefetch odd tile
        loadB(bN, kt + 1);
        mmas(aC, bC);               // compute even tile (partial wait only)
        if (kt + 2 < KT) {
            loadA(aC, kt + 2);      // prefetch next even tile
            loadB(bC, kt + 2);
        }
        mmas(aN, bN);               // compute odd tile
    }

    // C/D layout: VGPR r, lane l -> M = row0 + r + (l>=16 ? 8 : 0), col = l&15
#pragma unroll
    for (int i = 0; i < MT; ++i) {
        int rowb = (mg * MT + i) * 16 + ((lane >> 4) << 3);
#pragma unroll
        for (int j = 0; j < NT; ++j) {
            int col = (ng * NT + j) * 16 + (lane & 15);
            float bv = bias ? bias[col] : 0.0f;
#pragma unroll
            for (int r = 0; r < 8; ++r) {
                float v = acc[i][j][r] + bv;
                if (STORE_BF16)
                    ((bf16_t*)Cout)[(size_t)(rowb + r) * ldc + col] = (bf16_t)v;
                else
                    ((float*)Cout)[(size_t)(rowb + r) * ldc + col] = v;
            }
        }
    }
}

// ---------------------------------------------------------------------------
// LSTM gates: lin is B x 4*nn fp32 (pre-bias). Gate order: f, i, o | tanh(g).
// Writes new c (fp32 state) and new h as bf16 into up to two destinations.
// ---------------------------------------------------------------------------
__global__ void lstm_gate_kernel(const float* __restrict__ lin,
                                 const float* __restrict__ bias,
                                 float* __restrict__ c_state,
                                 bf16_t* __restrict__ hdst0, int ld0,
                                 bf16_t* __restrict__ hdst1, int ld1,
                                 int nn) {
    int idx = blockIdx.x * blockDim.x + threadIdx.x;
    if (idx >= BATCH * nn) return;
    int bi = idx / nn, n = idx - bi * nn;
    const float* L = lin + (size_t)bi * 4 * nn;
    float f = 1.0f / (1.0f + expf(-(L[n]          + bias[n])));
    float i = 1.0f / (1.0f + expf(-(L[nn + n]     + bias[nn + n])));
    float o = 1.0f / (1.0f + expf(-(L[2 * nn + n] + bias[2 * nn + n])));
    float g = tanhf(L[3 * nn + n] + bias[3 * nn + n]);
    float c = f * c_state[idx] + i * g;
    c_state[idx] = c;
    bf16_t hb = (bf16_t)(o * tanhf(c));
    hdst0[(size_t)bi * ld0 + n] = hb;
    if (hdst1) hdst1[(size_t)bi * ld1 + n] = hb;
}

// ---------------------------------------------------------------------------
// Initialize recurrent state buffers from the (fp32) h0/c0/h1/c1 inputs.
// ---------------------------------------------------------------------------
__global__ void init_state_kernel(const float* __restrict__ h0, const float* __restrict__ c0,
                                  const float* __restrict__ h1, const float* __restrict__ c1,
                                  bf16_t* __restrict__ h0b, float* __restrict__ c0b,
                                  bf16_t* __restrict__ h1b, float* __restrict__ c1b, int n) {
    int i = blockIdx.x * blockDim.x + threadIdx.x;
    if (i >= n) return;
    h0b[i] = (bf16_t)h0[i];
    h1b[i] = (bf16_t)h1[i];
    c0b[i] = c0[i];
    c1b[i] = c1[i];
}

// ---------------------------------------------------------------------------
extern "C" void kernel_launch(void* const* d_in, const int* in_sizes, int n_in,
                              void* d_out, int out_size, void* d_ws, size_t ws_size,
                              hipStream_t stream) {
    const float* in_inputs = (const float*)d_in[0];   // T,B,VOCAB
    const float* in_embw   = (const float*)d_in[1];   // VOCAB,EMB
    const float* in_w0     = (const float*)d_in[2];   // 1792,6144
    const float* in_b0     = (const float*)d_in[3];   // 6144
    const float* in_w1     = (const float*)d_in[4];   // 3072,6144
    const float* in_b1     = (const float*)d_in[5];   // 6144
    const float* in_outw   = (const float*)d_in[6];   // 1536,512
    const float* in_outb   = (const float*)d_in[7];   // 512
    const float* in_h0     = (const float*)d_in[8];
    const float* in_c0     = (const float*)d_in[9];
    const float* in_h1     = (const float*)d_in[10];
    const float* in_c1     = (const float*)d_in[11];

    const int K0L0 = EMBSZ, K1L0 = NN0;          // layer0: concat(emb_t, h0) = 1792
    const int K0L1 = NN0,   K1L1 = NN1;          // layer1: concat(h0n, h1)  = 3072
    const int N4_0 = 4 * NN0, N4_1 = 4 * NN1;    // 6144

    // ---- workspace layout (256 B aligned) ----
    char* ws = (char*)d_ws;
    size_t off = 0;
    auto alloc = [&](size_t bytes) -> char* {
        char* p = ws + off;
        off += (bytes + 255) & ~(size_t)255;
        return p;
    };
    bf16_t* w0p    = (bf16_t*)alloc((size_t)(K0L0 + K1L0) * N4_0 * 2);   // 22.0 MB
    bf16_t* w1p    = (bf16_t*)alloc((size_t)(K0L1 + K1L1) * N4_1 * 2);   // 37.7 MB
    bf16_t* embp   = (bf16_t*)alloc((size_t)VOCABSZ * EMBSZ * 2);
    bf16_t* outp   = (bf16_t*)alloc((size_t)NN1 * VOCABSZ * 2);
    bf16_t* in_bf  = (bf16_t*)alloc((size_t)T_STEPS * BATCH * VOCABSZ * 2); // 16.8 MB
    bf16_t* emb_bf = (bf16_t*)alloc((size_t)T_STEPS * BATCH * EMBSZ * 2);   //  8.4 MB
    bf16_t* rnn_bf = (bf16_t*)alloc((size_t)T_STEPS * BATCH * NN1 * 2);     // 50.3 MB
    float*  lin    = (float*) alloc((size_t)BATCH * N4_0 * 4);              //  1.6 MB
    bf16_t* h0buf  = (bf16_t*)alloc((size_t)BATCH * NN0 * 2);
    bf16_t* h1buf  = (bf16_t*)alloc((size_t)BATCH * NN1 * 2);
    float*  c0buf  = (float*) alloc((size_t)BATCH * NN0 * 4);
    float*  c1buf  = (float*) alloc((size_t)BATCH * NN1 * 4);
    (void)ws_size;

    auto cdiv = [](long a, long b) { return (int)((a + b - 1) / b); };
    const int BLK = 256;

    // ---- conversion / packing (must rerun every call: determinism) ----
    {
        long n = (long)T_STEPS * BATCH * VOCABSZ;
        f32_to_bf16_kernel<<<cdiv(n, BLK), BLK, 0, stream>>>(in_inputs, in_bf, (int)n);
    }
    pack_b_kernel<<<cdiv((long)VOCABSZ * EMBSZ, BLK), BLK, 0, stream>>>(in_embw, embp, VOCABSZ, EMBSZ);
    pack_b_kernel<<<cdiv((long)(K0L0 + K1L0) * N4_0, BLK), BLK, 0, stream>>>(in_w0, w0p, K0L0 + K1L0, N4_0);
    pack_b_kernel<<<cdiv((long)(K0L1 + K1L1) * N4_1, BLK), BLK, 0, stream>>>(in_w1, w1p, K0L1 + K1L1, N4_1);
    pack_b_kernel<<<cdiv((long)NN1 * VOCABSZ, BLK), BLK, 0, stream>>>(in_outw, outp, NN1, VOCABSZ);
    init_state_kernel<<<cdiv((long)BATCH * NN0, BLK), BLK, 0, stream>>>(
        in_h0, in_c0, in_h1, in_c1, h0buf, c0buf, h1buf, c1buf, BATCH * NN0);

    // ---- embedding GEMM: (T*B x VOCAB) @ (VOCAB x EMB) -> bf16 ----
    {
        int M = T_STEPS * BATCH;                          // 16384
        int waves = (M / 32) * ((EMBSZ / 16) / 2);        // 512*8 = 4096
        wmma_gemm_kernel<256, 2, 2, true><<<cdiv((long)waves * 32, BLK), BLK, 0, stream>>>(
            in_bf, VOCABSZ, VOCABSZ, in_bf, VOCABSZ, 0,
            embp, EMBSZ, nullptr, (void*)emb_bf, EMBSZ, M);
    }

    // ---- recurrent scan: 256 sequential steps ----
    // MT=2,NT=2: 2 m-groups x 192 n-groups = 384 waves; 64-thread blocks
    // (2 waves) so the 192 workgroups spread across many WGPs.
    const int LBLK = 64;
    int wavesL = (BATCH / 32) * ((N4_0 / 16) / 2);        // 2*192 = 384
    int gemmBlocksL = cdiv((long)wavesL * 32, LBLK);      // 192
    int gateBlocks  = cdiv((long)BATCH * NN0, BLK);       // 384
    for (int t = 0; t < T_STEPS; ++t) {
        const bf16_t* xt = emb_bf + (size_t)t * BATCH * EMBSZ;
        // layer 0: lin = [x_t | h0] @ w0
        wmma_gemm_kernel<64, 2, 2, false><<<gemmBlocksL, LBLK, 0, stream>>>(
            xt, EMBSZ, K0L0, h0buf, NN0, K1L0,
            w0p, N4_0, nullptr, (void*)lin, N4_0, BATCH);
        lstm_gate_kernel<<<gateBlocks, BLK, 0, stream>>>(
            lin, in_b0, c0buf, h0buf, NN0, nullptr, 0, NN0);
        // layer 1: lin = [h0n | h1] @ w1
        wmma_gemm_kernel<64, 2, 2, false><<<gemmBlocksL, LBLK, 0, stream>>>(
            h0buf, NN0, K0L1, h1buf, NN1, K1L1,
            w1p, N4_1, nullptr, (void*)lin, N4_1, BATCH);
        lstm_gate_kernel<<<gateBlocks, BLK, 0, stream>>>(
            lin, in_b1, c1buf, h1buf, NN1,
            rnn_bf + (size_t)t * BATCH * NN1, NN1, NN1);
    }

    // ---- output projection: (T*B x NN1) @ (NN1 x VOCAB) + out_b -> fp32 ----
    {
        int M = T_STEPS * BATCH;                          // 16384
        int waves = (M / 32) * ((VOCABSZ / 16) / 2);      // 512*16 = 8192
        wmma_gemm_kernel<256, 2, 2, false><<<cdiv((long)waves * 32, BLK), BLK, 0, stream>>>(
            rnn_bf, NN1, NN1, rnn_bf, NN1, 0,
            outp, VOCABSZ, in_outb, d_out, VOCABSZ, M);
    }
}
